// Model_1889785610620
// MI455X (gfx1250) — matrix operand order repaired and checked by reading, not compiled, not run on hardware
//
#include <hip/hip_runtime.h>
#include <hip/hip_bf16.h>
#include <math.h>

// ---------------------------------------------------------------------------
// Model dims
// ---------------------------------------------------------------------------
#define BB   128
#define SS   100
#define DD   512
#define NHH  8
#define DHH  64
#define HFF_ 2048
#define GHH  256
#define GLL  2
#define NCC  10
#define NEE  3
#define ROWS (BB * SS)      // 12800
#define SP   112            // S padded to 7*16 for WMMA tiles
#define SK   128            // padded K-dim for P@V (4 chunks of 32)
#define PLD  136            // LDS row stride (halves) for attention tiles

typedef __attribute__((ext_vector_type(16))) _Float16 f16x16;
typedef __attribute__((ext_vector_type(8)))  _Float16 f16x8;
typedef __attribute__((ext_vector_type(8)))  float    f32x8;
typedef __attribute__((ext_vector_type(4)))  unsigned int u32x4;
typedef __attribute__((ext_vector_type(8)))  int          i32x8;
typedef __attribute__((ext_vector_type(4)))  int          i32x4;

// Tensor Data Mover availability (device pass only; host pass takes fallback)
#if defined(__gfx1250__) && __has_builtin(__builtin_amdgcn_tensor_load_to_lds) \
    && __has_builtin(__builtin_amdgcn_s_wait_tensorcnt)
#define USE_TDM 1
#else
#define USE_TDM 0
#endif

#if USE_TDM
// DMA a 64-row x 32-half tile (row stride = strideElems halves) from global
// into LDS at ldsByteAddr, padding each 16-DWORD row with 8 DWORDs so rows
// land with stride 48 halves (96B) -- matching the [64][48] LDS tiles.
__device__ __forceinline__ void tdm_load_tile64x32(unsigned ldsByteAddr,
                                                   const _Float16* gptr,
                                                   int strideElems) {
  const unsigned long long va = (unsigned long long)(uintptr_t)gptr;
  u32x4 g0;
  g0[0] = 1u;                                   // count=1, user mode
  g0[1] = ldsByteAddr;                          // lds_addr (bytes)
  g0[2] = (unsigned)(va & 0xFFFFFFFFu);         // global_addr[31:0]
  g0[3] = (unsigned)((va >> 32) & 0x01FFFFFFu)  // global_addr[56:32]
          | (2u << 30);                         // type=2 ("image")
  i32x8 g1;
  g1[0] = (int)((1u << 16)      // data_size = 2 bytes
                | (1u << 20)    // pad_enable
                | (3u << 22)    // pad_interval: 16 DWORDs
                | (7u << 25));  // pad_amount: 8 DWORDs
  g1[1] = (int)(32u << 16);     // tensor_dim0 = 32 (low 16)
  g1[2] = (int)(64u << 16);     // tensor_dim0 hi=0 | tensor_dim1 = 64 (low 16)
  g1[3] = (int)(32u << 16);     // tensor_dim1 hi=0 | tile_dim0 = 32
  g1[4] = 64;                   // tile_dim1 = 64 | tile_dim2 = 0
  g1[5] = strideElems;          // tensor_dim0_stride[31:0]
  g1[6] = 0;                    // stride hi | tensor_dim1_stride lo
  g1[7] = 0;
  const i32x4 gz = {0, 0, 0, 0};
#if __clang_major__ >= 23
  const i32x8 gz8 = {0, 0, 0, 0, 0, 0, 0, 0};
  __builtin_amdgcn_tensor_load_to_lds(g0, g1, gz, gz, gz8, 0);
#else
  __builtin_amdgcn_tensor_load_to_lds(g0, g1, gz, gz, 0);
#endif
}
#endif

// ---------------------------------------------------------------------------
// Weight prep kernels
// ---------------------------------------------------------------------------
__global__ void convert_f16_kernel(const float* __restrict__ in,
                                   _Float16* __restrict__ out, int n) {
  for (int i = blockIdx.x * blockDim.x + threadIdx.x; i < n;
       i += gridDim.x * blockDim.x)
    out[i] = (_Float16)in[i];
}

// in: [R,C] f32 row-major  ->  out: [C,R] f16 row-major  (K-major weights)
__global__ void transpose_f16_kernel(const float* __restrict__ in,
                                     _Float16* __restrict__ out, int R, int C) {
  int n = R * C;
  for (int i = blockIdx.x * blockDim.x + threadIdx.x; i < n;
       i += gridDim.x * blockDim.x) {
    int r = i / C, c = i - r * C;
    out[c * R + r] = (_Float16)in[i];
  }
}

// ---------------------------------------------------------------------------
// Embedding gather: actF32[row][d] = emb[x[row]][d]; also f16 copy
// ---------------------------------------------------------------------------
__global__ void embed_kernel(const int* __restrict__ x,
                             const float* __restrict__ emb,
                             float* __restrict__ actF32,
                             _Float16* __restrict__ actF16) {
  int row = blockIdx.x;
  size_t tok = (size_t)x[row];
  const float* src = emb + tok * DD;
  for (int d = threadIdx.x; d < DD; d += blockDim.x) {
    float v = src[d];
    actF32[(size_t)row * DD + d] = v;
    actF16[(size_t)row * DD + d] = (_Float16)v;
  }
}

// ---------------------------------------------------------------------------
// WMMA GEMM:  out = act( A[M,K] @ B + bias + resid ),  B given as Bt[N][K].
// M%64==0, N%64==0, K%32==0.  256 threads = 8 waves; block tile 64x64;
// wave tile 16x32 (2 accumulators); double-buffered LDS fed by the
// Tensor Data Mover (fallback: register-staged copies).
// ---------------------------------------------------------------------------
template <bool RELU, bool RESID, bool OUTF32, bool OUTF16>
__global__ __launch_bounds__(256) void gemm_wmma_kernel(
    const _Float16* __restrict__ A, const _Float16* __restrict__ Bt,
    const float* __restrict__ bias, const float* __restrict__ resid,
    float* __restrict__ outF32, _Float16* __restrict__ outF16,
    int M, int N, int K) {
  __shared__ _Float16 As[2][64][48];   // [m][k], stride 96B (16B aligned)
  __shared__ _Float16 Bs[2][64][48];   // [n][k], stride 96B (16B aligned)

  const int tid = threadIdx.x;
  const int lane = tid & 31;
  const int wave = tid >> 5;
  const int halfsel = (lane >> 4) & 1;  // 0: lanes 0-15, 1: lanes 16-31
  const int l15 = lane & 15;

  const int bm = blockIdx.y * 64;
  const int bn = blockIdx.x * 64;
  const int wm = (wave & 3) * 16;       // wave row tile within block
  const int wn = (wave >> 2) * 32;      // wave col tile within block

  f32x8 acc0 = {0.f, 0.f, 0.f, 0.f, 0.f, 0.f, 0.f, 0.f};
  f32x8 acc1 = {0.f, 0.f, 0.f, 0.f, 0.f, 0.f, 0.f, 0.f};

  const int niter = K >> 5;

#if USE_TDM
  // wave 0 drives the DMA engine for the whole block
  if (tid < 32) {
    tdm_load_tile64x32((unsigned)(uintptr_t)&As[0][0][0],
                       A + (size_t)bm * K, K);
    tdm_load_tile64x32((unsigned)(uintptr_t)&Bs[0][0][0],
                       Bt + (size_t)bn * K, K);
    __builtin_amdgcn_s_wait_tensorcnt(0);
  }
  __syncthreads();
#else
  const int srow = tid >> 2;            // 0..63 (row of A tile / n of B tile)
  const int scol = (tid & 3) * 8;       // 0,8,16,24
  const _Float16* aSrc = A + (size_t)(bm + srow) * K + scol;
  const _Float16* bSrc = Bt + (size_t)(bn + srow) * K + scol;
  f16x8 aReg = *(const f16x8*)aSrc;
  f16x8 bReg = *(const f16x8*)bSrc;
  *(f16x8*)&As[0][srow][scol] = aReg;
  *(f16x8*)&Bs[0][srow][scol] = bReg;
  __syncthreads();
#endif

  for (int it = 0; it < niter; ++it) {
    const int cur = it & 1;
    const bool more = (it + 1) < niter;

#if USE_TDM
    if (more && tid < 32) {  // kick next-tile DMA; overlaps with the WMMAs
      const int k0 = (it + 1) << 5;
      tdm_load_tile64x32((unsigned)(uintptr_t)&As[cur ^ 1][0][0],
                         A + (size_t)bm * K + k0, K);
      tdm_load_tile64x32((unsigned)(uintptr_t)&Bs[cur ^ 1][0][0],
                         Bt + (size_t)bn * K + k0, K);
    }
#else
    if (more) {
      aReg = *(const f16x8*)(aSrc + (size_t)(it + 1) * 32);
      bReg = *(const f16x8*)(bSrc + (size_t)(it + 1) * 32);
    }
#endif

    // A fragment (16x32): lane<16 -> K 0..15 ; lane>=16 -> K 16..31
    f16x16 af;
    {
      const int kb = halfsel * 16;
      f16x8 lo = *(const f16x8*)&As[cur][wm + l15][kb];
      f16x8 hi = *(const f16x8*)&As[cur][wm + l15][kb + 8];
#pragma unroll
      for (int e = 0; e < 8; ++e) { af[e] = lo[e]; af[e + 8] = hi[e]; }
    }
    // B fragments (32x16): per-lane contiguous in K from [n][k] tile
    f16x16 b0, b1;
    {
      const int kb = halfsel * 16;
      f16x8 lo0 = *(const f16x8*)&Bs[cur][wn + l15][kb];
      f16x8 hi0 = *(const f16x8*)&Bs[cur][wn + l15][kb + 8];
      f16x8 lo1 = *(const f16x8*)&Bs[cur][wn + 16 + l15][kb];
      f16x8 hi1 = *(const f16x8*)&Bs[cur][wn + 16 + l15][kb + 8];
#pragma unroll
      for (int e = 0; e < 8; ++e) {
        b0[e] = lo0[e]; b0[e + 8] = hi0[e];
        b1[e] = lo1[e]; b1[e + 8] = hi1[e];
      }
    }
    acc0 = __builtin_amdgcn_wmma_f32_16x16x32_f16(false, af, false, b0,
                                                  (short)0, acc0, false, false);
    acc1 = __builtin_amdgcn_wmma_f32_16x16x32_f16(false, af, false, b1,
                                                  (short)0, acc1, false, false);

#if USE_TDM
    if (more && tid < 32) __builtin_amdgcn_s_wait_tensorcnt(0);
#else
    if (more) {
      *(f16x8*)&As[cur ^ 1][srow][scol] = aReg;
      *(f16x8*)&Bs[cur ^ 1][srow][scol] = bReg;
    }
#endif
    __syncthreads();
  }

  // epilogue; C/D layout: VGPR r -> row (r + halfsel*8), col = lane&15
  const int col0 = bn + wn + l15;
  const int col1 = col0 + 16;
  const float bias0 = bias[col0];
  const float bias1 = bias[col1];
  const int rbase = bm + wm + halfsel * 8;
#pragma unroll
  for (int r = 0; r < 8; ++r) {
    const int row = rbase + r;
    float v0 = acc0[r] + bias0;
    float v1 = acc1[r] + bias1;
    if (RESID) {
      v0 += resid[(size_t)row * N + col0];
      v1 += resid[(size_t)row * N + col1];
    }
    if (RELU) { v0 = fmaxf(v0, 0.f); v1 = fmaxf(v1, 0.f); }
    if (OUTF32) {
      outF32[(size_t)row * N + col0] = v0;
      outF32[(size_t)row * N + col1] = v1;
    }
    if (OUTF16) {
      outF16[(size_t)row * N + col0] = (_Float16)v0;
      outF16[(size_t)row * N + col1] = (_Float16)v1;
    }
  }
}

// ---------------------------------------------------------------------------
// Attention per (b,h): scores = softmax(Q K^T / 8), ctx = scores @ V.
// Contiguous-view head split: base = b*S*D + h*S*DH, row stride DH.
// Padded rows/cols are loaded via clamped indices (branch-free); the softmax
// phase overwrites or zeroes every padded entry before it is consumed.
// ---------------------------------------------------------------------------
__global__ __launch_bounds__(256) void attention_kernel(
    const _Float16* __restrict__ Q, const _Float16* __restrict__ K,
    const _Float16* __restrict__ V, _Float16* __restrict__ ctx) {
  __shared__ _Float16 Pf[SP][PLD];   // scores/probs, f16, 112x136
  __shared__ _Float16 Vt[DHH][PLD];  // V transposed [n][k], 64x136

  const int tid = threadIdx.x;
  const int lane = tid & 31;
  const int wave = tid >> 5;
  const int halfsel = (lane >> 4) & 1;
  const int l15 = lane & 15;

  const int bh = blockIdx.x;
  const int b = bh / NHH, h = bh % NHH;
  const size_t base = (size_t)b * SS * DD + (size_t)h * SS * DHH;
  const _Float16* Qp = Q + base;
  const _Float16* Kp = K + base;
  const _Float16* Vp = V + base;
  _Float16* Cp = ctx + base;

  // ---- stage Vt[n][k] = V[k][n]: wide global loads + LDS scatter
  if (tid < 2 * SS) {
    const int k = tid >> 1;        // 0..99
    const int p = (tid & 1) * 32;  // n half: 0 or 32
    _Float16 tmp[32];
#pragma unroll
    for (int j = 0; j < 4; ++j)
      *(f16x8*)&tmp[j * 8] = *(const f16x8*)(Vp + (size_t)k * DHH + p + j * 8);
#pragma unroll
    for (int j = 0; j < 32; ++j) Vt[p + j][k] = tmp[j];
  }
  for (int i = tid; i < DHH * (SK - SS); i += 256) {  // zero k = 100..127
    const int n = i / (SK - SS);
    const int k = SS + i % (SK - SS);
    Vt[n][k] = (_Float16)0.f;
  }

  // ---- phase 1: P = (Q @ K^T) * scale   (7x7 tiles of 16x16, K-dim = 64)
  for (int tile = wave; tile < 49; tile += 8) {
    const int ti = tile / 7, tj = tile % 7;
    f32x8 acc = {0.f, 0.f, 0.f, 0.f, 0.f, 0.f, 0.f, 0.f};
    const int qrow = ti * 16 + l15;
    const int qr = (qrow < SS) ? qrow : (SS - 1);   // clamp, no branch
    const int krow = tj * 16 + l15;
    const int kr = (krow < SS) ? krow : (SS - 1);   // clamp, no branch
#pragma unroll
    for (int kc = 0; kc < 64; kc += 32) {
      f16x16 af;
      {
        const int kb = halfsel * 16;
        f16x8 lo = *(const f16x8*)(Qp + (size_t)qr * DHH + kc + kb);
        f16x8 hi = *(const f16x8*)(Qp + (size_t)qr * DHH + kc + kb + 8);
#pragma unroll
        for (int e = 0; e < 8; ++e) { af[e] = lo[e]; af[e + 8] = hi[e]; }
      }
      f16x16 bf = *(const f16x16*)(Kp + (size_t)kr * DHH + kc + halfsel * 16);
      acc = __builtin_amdgcn_wmma_f32_16x16x32_f16(false, af, false, bf,
                                                   (short)0, acc, false, false);
    }
#pragma unroll
    for (int r = 0; r < 8; ++r)
      Pf[ti * 16 + halfsel * 8 + r][tj * 16 + l15] =
          (_Float16)(acc[r] * 0.125f);
  }
  __syncthreads();

  // ---- phase 2: wave-parallel row softmax over first SS columns
  for (int row = wave; row < SS; row += 8) {
    float v[4];
#pragma unroll
    for (int i = 0; i < 4; ++i) {
      const int c = lane + i * 32;
      v[i] = (c < SS) ? (float)Pf[row][c] : -3.0e38f;
    }
    float m = fmaxf(fmaxf(v[0], v[1]), fmaxf(v[2], v[3]));
#pragma unroll
    for (int off = 16; off > 0; off >>= 1)
      m = fmaxf(m, __shfl_xor(m, off, 32));
    float e[4];
    float s = 0.f;
#pragma unroll
    for (int i = 0; i < 4; ++i) {
      const int c = lane + i * 32;
      e[i] = (c < SS) ? __expf(v[i] - m) : 0.f;
      s += e[i];
    }
#pragma unroll
    for (int off = 16; off > 0; off >>= 1) s += __shfl_xor(s, off, 32);
    const float inv = 1.f / s;
#pragma unroll
    for (int i = 0; i < 4; ++i) {
      const int c = lane + i * 32;
      Pf[row][c] = (_Float16)(e[i] * inv);  // cols >= SS get exact 0
    }
  }
  // zero the padding rows so they contribute nothing as A-matrix rows
  for (int i = tid; i < (SP - SS) * SK; i += 256) {
    const int r = SS + i / SK, c = i % SK;
    Pf[r][c] = (_Float16)0.f;
  }
  __syncthreads();

  // ---- phase 3: ctx = P[112,128] @ V[128(pad),64]   (7x4 tiles, 4 K-chunks)
  for (int tile = wave; tile < 28; tile += 8) {
    const int ti = tile / 4, tj = tile % 4;
    f32x8 acc = {0.f, 0.f, 0.f, 0.f, 0.f, 0.f, 0.f, 0.f};
#pragma unroll
    for (int kc = 0; kc < SK; kc += 32) {
      const int prow = ti * 16 + l15;
      f16x16 af;
      {
        const int kb = halfsel * 16;
        f16x8 lo = *(const f16x8*)&Pf[prow][kc + kb];
        f16x8 hi = *(const f16x8*)&Pf[prow][kc + kb + 8];
#pragma unroll
        for (int e = 0; e < 8; ++e) { af[e] = lo[e]; af[e + 8] = hi[e]; }
      }
      const int vcol = tj * 16 + l15;
      f16x16 bf;
      {
        const int kb = kc + halfsel * 16;
        f16x8 lo = *(const f16x8*)&Vt[vcol][kb];
        f16x8 hi = *(const f16x8*)&Vt[vcol][kb + 8];
#pragma unroll
        for (int e = 0; e < 8; ++e) { bf[e] = lo[e]; bf[e + 8] = hi[e]; }
      }
      acc = __builtin_amdgcn_wmma_f32_16x16x32_f16(false, af, false, bf,
                                                   (short)0, acc, false, false);
    }
#pragma unroll
    for (int r = 0; r < 8; ++r) {
      const int row = ti * 16 + halfsel * 8 + r;
      if (row < SS)
        Cp[(size_t)row * DHH + tj * 16 + l15] = (_Float16)acc[r];
    }
  }
}

// ---------------------------------------------------------------------------
// LayerNorm over D=512, one block per row; writes f32 + f16
// ---------------------------------------------------------------------------
__global__ __launch_bounds__(256) void layernorm_kernel(
    const float* __restrict__ in, const float* __restrict__ g,
    const float* __restrict__ be, float* __restrict__ outF32,
    _Float16* __restrict__ outF16) {
  __shared__ float s1[256], s2[256];
  __shared__ float mean_s, rstd_s;
  const int row = blockIdx.x, tid = threadIdx.x;
  const float* r = in + (size_t)row * DD;
  const float a = r[tid], b = r[tid + 256];
  s1[tid] = a + b;
  s2[tid] = a * a + b * b;
  __syncthreads();
  for (int s = 128; s > 0; s >>= 1) {
    if (tid < s) { s1[tid] += s1[tid + s]; s2[tid] += s2[tid + s]; }
    __syncthreads();
  }
  if (tid == 0) {
    float m = s1[0] * (1.f / DD);
    float v = s2[0] * (1.f / DD) - m * m;
    mean_s = m;
    rstd_s = rsqrtf(v + 1e-5f);
  }
  __syncthreads();
  const float m = mean_s, rs = rstd_s;
  float o0 = (a - m) * rs * g[tid] + be[tid];
  float o1 = (b - m) * rs * g[tid + 256] + be[tid + 256];
  outF32[(size_t)row * DD + tid] = o0;
  outF32[(size_t)row * DD + tid + 256] = o1;
  outF16[(size_t)row * DD + tid] = (_Float16)o0;
  outF16[(size_t)row * DD + tid + 256] = (_Float16)o1;
}

// ---------------------------------------------------------------------------
// (B,S,D) f32 -> (S,B,D) f16 for GRU input
// ---------------------------------------------------------------------------
__global__ void transpose_act_kernel(const float* __restrict__ in,
                                     _Float16* __restrict__ out) {
  const int total = SS * BB * DD;
  for (int i = blockIdx.x * blockDim.x + threadIdx.x; i < total;
       i += gridDim.x * blockDim.x) {
    int s = i / (BB * DD);
    int rem = i - s * (BB * DD);
    int b = rem / DD;
    int d = rem - b * DD;
    out[i] = (_Float16)in[((size_t)b * SS + s) * DD + d];
  }
}

// concat ys0/ys1 [S,B,GH] f32 -> hseq [S,B,2GH] f16
__global__ void concat_kernel(const float* __restrict__ ys0,
                              const float* __restrict__ ys1,
                              _Float16* __restrict__ out) {
  const int total = SS * BB * (2 * GHH);
  for (int i = blockIdx.x * blockDim.x + threadIdx.x; i < total;
       i += gridDim.x * blockDim.x) {
    int sb = i / (2 * GHH);
    int j = i - sb * (2 * GHH);
    float v = (j < GHH) ? ys0[(size_t)sb * GHH + j]
                        : ys1[(size_t)sb * GHH + (j - GHH)];
    out[i] = (_Float16)v;
  }
}

// ---------------------------------------------------------------------------
// GRU gate math for one timestep: grid BB blocks x GHH threads
// ---------------------------------------------------------------------------
__global__ __launch_bounds__(GHH) void gru_gate_kernel(
    const float* __restrict__ xp_t, const float* __restrict__ hp,
    const float* __restrict__ hprev, float* __restrict__ ys_t,
    _Float16* __restrict__ hF16) {
  const int b = blockIdx.x, j = threadIdx.x;
  const float* xr = xp_t + (size_t)b * (3 * GHH);
  const float* hr = hp + (size_t)b * (3 * GHH);
  const float rg = 1.f / (1.f + __expf(-(xr[j] + hr[j])));
  const float zg = 1.f / (1.f + __expf(-(xr[GHH + j] + hr[GHH + j])));
  const float ng = tanhf(xr[2 * GHH + j] + rg * hr[2 * GHH + j]);
  const float h = hprev ? hprev[(size_t)b * GHH + j] : 0.f;
  const float hn = (1.f - zg) * ng + zg * h;
  ys_t[(size_t)b * GHH + j] = hn;
  hF16[(size_t)b * GHH + j] = (_Float16)hn;
}

// ---------------------------------------------------------------------------
// feat = relu(concat(embed_feat, gru_out)); pooled = max over seq
// ---------------------------------------------------------------------------
__global__ __launch_bounds__(256) void pool_kernel(
    const float* __restrict__ act, const float* __restrict__ ys0,
    const float* __restrict__ ys1, float* __restrict__ pooled) {
  const int b = blockIdx.x;
  for (int f = threadIdx.x; f < DD + 2 * GHH; f += 256) {
    float m = -3.0e38f;
    for (int s = 0; s < SS; ++s) {
      float v;
      if (f < DD)
        v = act[((size_t)b * SS + s) * DD + f];
      else if (f < DD + GHH)
        v = ys0[((size_t)s * BB + b) * GHH + (f - DD)];
      else
        v = ys1[((size_t)s * BB + b) * GHH + (f - DD - GHH)];
      m = fmaxf(m, v);
    }
    pooled[(size_t)b * (DD + 2 * GHH) + f] = fmaxf(m, 0.f);
  }
}

// out[b][c] = pooled[b] . fc_W[:,c] + fc_b[c]
__global__ __launch_bounds__(256) void fc_kernel(const float* __restrict__ pooled,
                                                 const float* __restrict__ W,
                                                 const float* __restrict__ bias,
                                                 float* __restrict__ out) {
  __shared__ float red[256];
  const int b = blockIdx.x, tid = threadIdx.x;
  const int F = DD + 2 * GHH;
  float acc[NCC];
#pragma unroll
  for (int c = 0; c < NCC; ++c) acc[c] = 0.f;
  for (int f = tid; f < F; f += 256) {
    const float p = pooled[(size_t)b * F + f];
#pragma unroll
    for (int c = 0; c < NCC; ++c) acc[c] += p * W[(size_t)f * NCC + c];
  }
  for (int c = 0; c < NCC; ++c) {
    red[tid] = acc[c];
    __syncthreads();
    for (int s = 128; s > 0; s >>= 1) {
      if (tid < s) red[tid] += red[tid + s];
      __syncthreads();
    }
    if (tid == 0) out[b * NCC + c] = red[0] + bias[c];
    __syncthreads();
  }
}

// ---------------------------------------------------------------------------
// Host launcher
// ---------------------------------------------------------------------------
extern "C" void kernel_launch(void* const* d_in, const int* in_sizes, int n_in,
                              void* d_out, int out_size, void* d_ws,
                              size_t ws_size, hipStream_t stream) {
  (void)in_sizes; (void)n_in; (void)out_size; (void)ws_size;

  const int*   x       = (const int*)d_in[0];
  const float* emb     = (const float*)d_in[2];
  const float* Wq      = (const float*)d_in[3];
  const float* bq      = (const float*)d_in[4];
  const float* Wk      = (const float*)d_in[5];
  const float* bk      = (const float*)d_in[6];
  const float* Wv      = (const float*)d_in[7];
  const float* bv      = (const float*)d_in[8];
  const float* Wo      = (const float*)d_in[9];
  const float* bo      = (const float*)d_in[10];
  const float* g1      = (const float*)d_in[11];
  const float* be1     = (const float*)d_in[12];
  const float* W1      = (const float*)d_in[13];
  const float* b1      = (const float*)d_in[14];
  const float* W2      = (const float*)d_in[15];
  const float* b2      = (const float*)d_in[16];
  const float* g2      = (const float*)d_in[17];
  const float* be2     = (const float*)d_in[18];
  const float* gruWih  = (const float*)d_in[19];
  const float* gruWhh  = (const float*)d_in[20];
  const float* gruBih  = (const float*)d_in[21];
  const float* gruBhh  = (const float*)d_in[22];
  const float* fcW     = (const float*)d_in[23];
  const float* fcB     = (const float*)d_in[24];
  float* out = (float*)d_out;

  // --- workspace carve-out (~220 MB) ---
  char* ws = (char*)d_ws;
  size_t off = 0;
  auto alloc = [&](size_t bytes) -> void* {
    void* p = ws + off;
    off = (off + bytes + 255) & ~(size_t)255;
    return p;
  };
  float*     actF32  = (float*)alloc((size_t)ROWS * DD * 4);
  _Float16*  actF16  = (_Float16*)alloc((size_t)ROWS * DD * 2);
  float*     tmpF32  = (float*)alloc((size_t)ROWS * DD * 4);
  _Float16*  bufF16  = (_Float16*)alloc((size_t)ROWS * HFF_ * 2); // q|k|v|ctx or ff1
  _Float16*  qF16    = bufF16;
  _Float16*  kF16    = bufF16 + (size_t)ROWS * DD;
  _Float16*  vF16    = bufF16 + (size_t)2 * ROWS * DD;
  _Float16*  ctxF16  = bufF16 + (size_t)3 * ROWS * DD;
  _Float16*  hseqF16 = (_Float16*)alloc((size_t)SS * BB * DD * 2);
  float*     xpF32   = (float*)alloc((size_t)ROWS * 3 * GHH * 4);
  float*     hpF32   = (float*)alloc((size_t)BB * 3 * GHH * 4);
  float*     ys0     = (float*)alloc((size_t)SS * BB * GHH * 4);
  float*     ys1     = (float*)alloc((size_t)SS * BB * GHH * 4);
  _Float16*  hF16    = (_Float16*)alloc((size_t)BB * GHH * 2);
  float*     pooled  = (float*)alloc((size_t)BB * (DD + 2 * GHH) * 4);
  // weights, all stored as Bt[N][K] f16
  _Float16*  WqT     = (_Float16*)alloc((size_t)NEE * DD * DD * 2);
  _Float16*  WkT     = (_Float16*)alloc((size_t)NEE * DD * DD * 2);
  _Float16*  WvT     = (_Float16*)alloc((size_t)NEE * DD * DD * 2);
  _Float16*  WoT     = (_Float16*)alloc((size_t)NEE * DD * DD * 2);
  _Float16*  W1T     = (_Float16*)alloc((size_t)NEE * DD * HFF_ * 2);
  _Float16*  W2T     = (_Float16*)alloc((size_t)NEE * HFF_ * DD * 2);
  _Float16*  WihN    = (_Float16*)alloc((size_t)GLL * 2 * 3 * GHH * DD * 2);
  _Float16*  WhhN    = (_Float16*)alloc((size_t)GLL * 2 * 3 * GHH * GHH * 2);

  // --- weight prep (once per call) ---
  for (int l = 0; l < NEE; ++l) {
    const size_t o = (size_t)l * DD * DD;
    transpose_f16_kernel<<<512, 256, 0, stream>>>(Wq + o, WqT + o, DD, DD);
    transpose_f16_kernel<<<512, 256, 0, stream>>>(Wk + o, WkT + o, DD, DD);
    transpose_f16_kernel<<<512, 256, 0, stream>>>(Wv + o, WvT + o, DD, DD);
    transpose_f16_kernel<<<512, 256, 0, stream>>>(Wo + o, WoT + o, DD, DD);
    const size_t of = (size_t)l * DD * HFF_;
    transpose_f16_kernel<<<2048, 256, 0, stream>>>(W1 + of, W1T + of, DD, HFF_);
    transpose_f16_kernel<<<2048, 256, 0, stream>>>(W2 + of, W2T + of, HFF_, DD);
  }
  // GRU weights are [3GH, Din] / [3GH, GH] == [N][K] already: plain convert
  convert_f16_kernel<<<2048, 256, 0, stream>>>(gruWih, WihN,
                                               GLL * 2 * 3 * GHH * DD);
  convert_f16_kernel<<<1024, 256, 0, stream>>>(gruWhh, WhhN,
                                               GLL * 2 * 3 * GHH * GHH);

  // --- embedding ---
  embed_kernel<<<ROWS, 128, 0, stream>>>(x, emb, actF32, actF16);

  // GEMM variant launchers (Bt is [N][K])
  auto gemmF16 = [&](const _Float16* A, const _Float16* Bt, const float* bias,
                     _Float16* oF16, int M, int N, int K) {
    dim3 grid(N / 64, M / 64);
    gemm_wmma_kernel<false, false, false, true><<<grid, 256, 0, stream>>>(
        A, Bt, bias, nullptr, nullptr, oF16, M, N, K);
  };
  auto gemmReluF16 = [&](const _Float16* A, const _Float16* Bt,
                         const float* bias, _Float16* oF16, int M, int N,
                         int K) {
    dim3 grid(N / 64, M / 64);
    gemm_wmma_kernel<true, false, false, true><<<grid, 256, 0, stream>>>(
        A, Bt, bias, nullptr, nullptr, oF16, M, N, K);
  };
  auto gemmResidF32 = [&](const _Float16* A, const _Float16* Bt,
                          const float* bias, const float* resid, float* oF32,
                          int M, int N, int K) {
    dim3 grid(N / 64, M / 64);
    gemm_wmma_kernel<false, true, true, false><<<grid, 256, 0, stream>>>(
        A, Bt, bias, resid, oF32, nullptr, M, N, K);
  };
  auto gemmF32 = [&](const _Float16* A, const _Float16* Bt, const float* bias,
                     float* oF32, int M, int N, int K) {
    dim3 grid(N / 64, M / 64);
    gemm_wmma_kernel<false, false, true, false><<<grid, 256, 0, stream>>>(
        A, Bt, bias, nullptr, oF32, nullptr, M, N, K);
  };

  // --- encoder stack ---
  for (int l = 0; l < NEE; ++l) {
    const size_t wOff = (size_t)l * DD * DD;
    gemmF16(actF16, WqT + wOff, bq + l * DD, qF16, ROWS, DD, DD);
    gemmF16(actF16, WkT + wOff, bk + l * DD, kF16, ROWS, DD, DD);
    gemmF16(actF16, WvT + wOff, bv + l * DD, vF16, ROWS, DD, DD);
    attention_kernel<<<BB * NHH, 256, 0, stream>>>(qF16, kF16, vF16, ctxF16);
    gemmResidF32(ctxF16, WoT + wOff, bo + l * DD, actF32, tmpF32, ROWS, DD, DD);
    layernorm_kernel<<<ROWS, 256, 0, stream>>>(tmpF32, g1 + l * DD,
                                               be1 + l * DD, actF32, actF16);
    gemmReluF16(actF16, W1T + (size_t)l * DD * HFF_, b1 + l * HFF_, bufF16,
                ROWS, HFF_, DD);
    gemmResidF32(bufF16, W2T + (size_t)l * HFF_ * DD, b2 + l * DD, actF32,
                 tmpF32, ROWS, DD, HFF_);
    layernorm_kernel<<<ROWS, 256, 0, stream>>>(tmpF32, g2 + l * DD,
                                               be2 + l * DD, actF32, actF16);
  }

  // --- GRU stack (seq-major) ---
  transpose_act_kernel<<<8192, 256, 0, stream>>>(actF32, hseqF16);
  for (int l = 0; l < GLL; ++l) {
    for (int d = 0; d < 2; ++d) {
      const int ld = l * 2 + d;
      float* ys = (d == 0) ? ys0 : ys1;
      gemmF32(hseqF16, WihN + (size_t)ld * 3 * GHH * DD, gruBih + ld * 3 * GHH,
              xpF32, ROWS, 3 * GHH, DD);
      hipMemsetAsync(hF16, 0, (size_t)BB * GHH * 2, stream);
      for (int step = 0; step < SS; ++step) {
        const int t = (d == 0) ? step : (SS - 1 - step);
        gemmF32(hF16, WhhN + (size_t)ld * 3 * GHH * GHH, gruBhh + ld * 3 * GHH,
                hpF32, BB, 3 * GHH, GHH);
        const float* hprev =
            (step == 0) ? nullptr
                        : ys + (size_t)((d == 0) ? (t - 1) : (t + 1)) * BB * GHH;
        gru_gate_kernel<<<BB, GHH, 0, stream>>>(
            xpF32 + (size_t)t * BB * 3 * GHH, hpF32, hprev,
            ys + (size_t)t * BB * GHH, hF16);
      }
    }
    if (l == 0)
      concat_kernel<<<8192, 256, 0, stream>>>(ys0, ys1, hseqF16);
  }

  // --- pool + classifier ---
  pool_kernel<<<BB, 256, 0, stream>>>(actF32, ys0, ys1, pooled);
  fc_kernel<<<BB, 256, 0, stream>>>(pooled, fcW, fcB, out);
}
